// dLSTM_58445914964282
// MI455X (gfx1250) — compile-verified, matrix operand order checked by reading, not run on hardware
//
#include <hip/hip_runtime.h>

// ---------------- problem constants ----------------
constexpr int T = 16, B = 32, H = 512, V = 32000, S = 20;
constexpr int R  = T * B;        // 512 rows, row = t*B + b
constexpr int H2 = 2 * H;        // 1024
constexpr int H3 = 3 * H;        // 1536  (K of both GEMMs)
constexpr int H4 = 4 * H;        // 2048  (gates N)

constexpr int CH  = 64;          // K-chunk staged per TDM transfer (elements)
constexpr int STR = 72;          // padded LDS row stride (elems) = 36 dwords

typedef __attribute__((ext_vector_type(16))) __bf16 v16bf;
typedef __attribute__((ext_vector_type(8)))  float  v8f;
typedef __attribute__((ext_vector_type(4)))  unsigned int u32x4;
typedef __attribute__((ext_vector_type(8)))  int  i32x8;
typedef __attribute__((ext_vector_type(4)))  int  i32x4;

// ---------------- helpers ----------------
__device__ __forceinline__ unsigned short f2bf(float f) {
    unsigned int u = __float_as_uint(f);
    unsigned int r = u + 0x7FFFu + ((u >> 16) & 1u);   // round-to-nearest-even
    return (unsigned short)(r >> 16);
}

__device__ __forceinline__ float sigf(float x) { return 1.0f / (1.0f + expf(-x)); }

// 16x32 bf16 fragment from LDS-staged tile (row stride = STR, rows of 64 elems)
__device__ __forceinline__ v16bf lds_frag(const unsigned short* sh, int rowBase, int ksub) {
    int lane = threadIdx.x & 31;
    int half = lane >> 4;
    int r    = lane & 15;
    const unsigned short* p = sh + (rowBase + r) * STR + ksub + half * 8;
    union { v16bf v; int4 q[2]; } u;
    u.q[0] = *(const int4*)(p);
    u.q[1] = *(const int4*)(p + 16);
    return u.v;
}

__device__ __forceinline__ unsigned lds_off(const void* p) {
    return (unsigned)(unsigned long long)p;   // flat LDS aperture: low 32b = LDS offset
}

// TDM: async-load a [rows x 64] bf16 tile (global row stride 1536 elems) into LDS,
// padding +4 dwords after every 32-dword row -> LDS row stride 144B (bank spread).
__device__ __forceinline__ void tdm_load_2d(unsigned ldsOff, const void* gptr, int rows) {
    unsigned long long ga = (unsigned long long)gptr;
    u32x4 g0;
    g0[0] = 1u;                                            // count=1 (valid), load, user
    g0[1] = ldsOff;                                        // lds_addr
    g0[2] = (unsigned)ga;                                  // global_addr[31:0]
    g0[3] = (unsigned)((ga >> 32) & 0x01FFFFFFull) | (2u << 30);  // addr[56:32] | type=2
    i32x8 g1;
    g1[0] = (1 << 16) | (1 << 20) | (4 << 22) | (3 << 25); // 2B elems, pad 4dw per 32dw
    g1[1] = (int)((1536u & 0xFFFFu) << 16);                // tensor_dim0 = 1536
    g1[2] = (int)((32000u & 0xFFFFu) << 16);               // tensor_dim1 (OOB bound)
    g1[3] = (int)(64u << 16);                              // tile_dim0 = 64 (K elems)
    g1[4] = rows;                                          // tile_dim1 = rows
    g1[5] = 1536;                                          // tensor_dim0_stride
    g1[6] = 0;
    g1[7] = 0;
    i32x4 z4 = {0, 0, 0, 0};
#if defined(__clang_major__) && __clang_major__ >= 23
    i32x8 z8 = {0, 0, 0, 0, 0, 0, 0, 0};
    __builtin_amdgcn_tensor_load_to_lds(g0, g1, z4, z4, z8, 0);
#else
    __builtin_amdgcn_tensor_load_to_lds(g0, g1, z4, z4, 0);
#endif
}

// ---------------- one-time prep kernels ----------------
__global__ __launch_bounds__(256) void k_convert_wlin(const float* __restrict__ src,
                                                      unsigned short* __restrict__ dst,
                                                      long n) {
    long i = ((long)blockIdx.x * 256 + threadIdx.x) * 8;
    if (i + 7 >= n) return;
    const float4* s = (const float4*)(src + i);
    float4 x = s[0], y = s[1];
    uint4 o = make_uint4(
        (unsigned)f2bf(x.x) | ((unsigned)f2bf(x.y) << 16),
        (unsigned)f2bf(x.z) | ((unsigned)f2bf(x.w) << 16),
        (unsigned)f2bf(y.x) | ((unsigned)f2bf(y.y) << 16),
        (unsigned)f2bf(y.z) | ((unsigned)f2bf(y.w) << 16));
    *(uint4*)(dst + i) = o;
}

__global__ __launch_bounds__(256) void k_build_wcat(const float* __restrict__ W_ih,
                                                    const float* __restrict__ W_hh,
                                                    const float* __restrict__ b_ih,
                                                    const float* __restrict__ b_hh,
                                                    unsigned short* __restrict__ Wcat,
                                                    float* __restrict__ bias) {
    int idx = blockIdx.x * 256 + threadIdx.x;
    if (idx >= H4 * H3) return;
    int g = idx / H3, c = idx % H3;
    float v = (c < H) ? W_hh[g * H + c] : W_ih[g * H2 + (c - H)];
    Wcat[idx] = f2bf(v);
    if (idx < H4) bias[idx] = b_ih[idx] + b_hh[idx];
}

__global__ __launch_bounds__(256) void k_init(const float* __restrict__ temb,
                                              unsigned short* __restrict__ Abuf,
                                              float* __restrict__ cx,
                                              float* __restrict__ loss,
                                              int* __restrict__ w_i) {
    int idx = blockIdx.x * 256 + threadIdx.x;
    if (idx >= R * H) return;
    int r = idx / H, h = idx % H;
    cx[idx] = 0.0f;
    Abuf[r * H3 + h] = 0;                                 // hx = 0
    Abuf[r * H3 + H + h] = f2bf(temb[(r >> 5) * H + h]);  // t = r / B
    if (h == 0) { loss[r] = 0.0f; w_i[r] = 0; }
}

// ---------------- per-step kernels ----------------
__global__ __launch_bounds__(256) void k_embed(const float* __restrict__ wemb,
                                               const int* __restrict__ w_i,
                                               unsigned short* __restrict__ Abuf) {
    int idx = blockIdx.x * 256 + threadIdx.x;
    if (idx >= R * H) return;
    int r = idx / H, h = idx % H;
    Abuf[r * H3 + H2 + h] = f2bf(wemb[(long)w_i[r] * H + h]);
}

// gates[512,2048] = Abuf[512,1536] @ Wcat^T  -- TDM double-buffered, LDS-fed WMMA
__global__ __launch_bounds__(256) void k_gates_gemm(const unsigned short* __restrict__ A,
                                                    const unsigned short* __restrict__ W,
                                                    float* __restrict__ C) {
    __shared__ unsigned short shW[2][128 * STR];   // 2 x 18432 B
    __shared__ unsigned short shA[2][64 * STR];    // 2 x  9216 B

    int waveId = threadIdx.x >> 5;
    int rw = waveId & 3, cw = waveId >> 2;
    int row0 = blockIdx.y * 64;
    int col0 = blockIdx.x * 128;
    int rowLoc = rw * 16, colLoc = cw * 64;

    const unsigned short* Ag = A + (long)row0 * H3;
    const unsigned short* Wg = W + (long)col0 * H3;

    v8f zero = {0.f, 0.f, 0.f, 0.f, 0.f, 0.f, 0.f, 0.f};
    v8f acc[4] = {zero, zero, zero, zero};

    constexpr int NCH = H3 / CH;   // 24

    if (threadIdx.x < 32) {
        tdm_load_2d(lds_off(&shA[0][0]), Ag, 64);
        tdm_load_2d(lds_off(&shW[0][0]), Wg, 128);
    }
    for (int c = 0; c < NCH; ++c) {
        int buf = c & 1;
        if (c + 1 < NCH) {
            if (threadIdx.x < 32) {
                tdm_load_2d(lds_off(&shA[buf ^ 1][0]), Ag + (c + 1) * CH, 64);
                tdm_load_2d(lds_off(&shW[buf ^ 1][0]), Wg + (c + 1) * CH, 128);
                __builtin_amdgcn_s_wait_tensorcnt(2);   // chunk c's pair retired
            }
        } else {
            if (threadIdx.x < 32) __builtin_amdgcn_s_wait_tensorcnt(0);
        }
        __syncthreads();
#pragma unroll
        for (int ksub = 0; ksub < CH; ksub += 32) {
            // batch all fragment loads first -> one ds wait, then WMMA burst
            v16bf a = lds_frag(shA[buf], rowLoc, ksub);
            v16bf b[4];
#pragma unroll
            for (int t = 0; t < 4; ++t)
                b[t] = lds_frag(shW[buf], colLoc + t * 16, ksub);
#pragma unroll
            for (int t = 0; t < 4; ++t)
                acc[t] = __builtin_amdgcn_wmma_f32_16x16x32_bf16(
                    false, a, false, b[t], (short)0, acc[t], false, false);
        }
        __syncthreads();
    }
    int lane = threadIdx.x & 31, half = lane >> 4, r = lane & 15;
#pragma unroll
    for (int t = 0; t < 4; ++t)
#pragma unroll
        for (int j = 0; j < 8; ++j)
            C[(long)(row0 + rowLoc + half * 8 + j) * H4 + col0 + colLoc + t * 16 + r] = acc[t][j];
}

__global__ __launch_bounds__(256) void k_cell(const float* __restrict__ gates,
                                              const float* __restrict__ bias,
                                              float* __restrict__ cx,
                                              unsigned short* __restrict__ Abuf) {
    int idx = blockIdx.x * 256 + threadIdx.x;
    if (idx >= R * H) return;
    int r = idx / H, h = idx % H;
    const float* g = gates + (long)r * H4;
    float ig = g[h]          + bias[h];
    float fg = g[H + h]      + bias[H + h];
    float gg = g[H2 + h]     + bias[H2 + h];
    float og = g[3 * H + h]  + bias[3 * H + h];
    float c  = sigf(fg) * cx[idx] + sigf(ig) * tanhf(gg);
    float hv = sigf(og) * tanhf(c);
    cx[idx] = c;
    Abuf[r * H3 + h] = f2bf(hv);
}

__global__ __launch_bounds__(256) void k_zero_partials(float* __restrict__ sumexp,
                                                       unsigned long long* __restrict__ argkey,
                                                       float* __restrict__ tgtval) {
    int r = blockIdx.x * 256 + threadIdx.x;
    if (r >= R) return;
    sumexp[r] = 0.0f; argkey[r] = 0ull; tgtval[r] = 0.0f;
}

// logits[512,32000] = tanh(Abuf @ Wlin^T + b_lin).
// Block = 128 rows x 128 cols; wave = 16 rows x 128 cols (8 accumulators).
// W double-buffered + A single-buffered TDM staging; in-order TENSORcnt
// retirement lets s_wait_tensorcnt(1) cover both W(c) and A(c).
__global__ __launch_bounds__(256) void k_logits(const unsigned short* __restrict__ A,
                                                const unsigned short* __restrict__ Wlin,
                                                const float* __restrict__ b_lin,
                                                const int* __restrict__ target, int s,
                                                float* __restrict__ sumexp,
                                                unsigned long long* __restrict__ argkey,
                                                float* __restrict__ tgtval) {
    __shared__ unsigned short shW[2][128 * STR];   // 2 x 18432 B
    __shared__ unsigned short shA[128 * STR];      //     18432 B  (55.3 KB total)

    int waveId = threadIdx.x >> 5;                 // wave w: rows [w*16, w*16+16)
    int row0 = blockIdx.y * 128;
    int col0 = blockIdx.x * 128;
    int rowLoc = waveId * 16;

    const unsigned short* Ag = A    + (long)row0 * H3;
    const unsigned short* Wg = Wlin + (long)col0 * H3;

    v8f zero = {0.f, 0.f, 0.f, 0.f, 0.f, 0.f, 0.f, 0.f};
    v8f acc[8] = {zero, zero, zero, zero, zero, zero, zero, zero};

    constexpr int NCH = H3 / CH;   // 24

    // prologue: issue order W0, A0 (in-order completion per wave)
    if (threadIdx.x < 32) {
        tdm_load_2d(lds_off(&shW[0][0]), Wg, 128);
        tdm_load_2d(lds_off(&shA[0]),    Ag, 128);
    }
    for (int c = 0; c < NCH; ++c) {
        int buf = c & 1;
        if (c + 1 < NCH) {
            if (threadIdx.x < 32) {
                tdm_load_2d(lds_off(&shW[buf ^ 1][0]), Wg + (c + 1) * CH, 128);
                __builtin_amdgcn_s_wait_tensorcnt(1);   // ... W(c), A(c) retired
            }
        } else {
            if (threadIdx.x < 32) __builtin_amdgcn_s_wait_tensorcnt(0);
        }
        __syncthreads();                                // publish chunk c
#pragma unroll
        for (int ksub = 0; ksub < CH; ksub += 32) {
            // batch all fragment loads first -> one ds wait, then 8-deep WMMA burst
            v16bf a = lds_frag(shA, rowLoc, ksub);
            v16bf b[8];
#pragma unroll
            for (int t = 0; t < 8; ++t)
                b[t] = lds_frag(shW[buf], t * 16, ksub);
#pragma unroll
            for (int t = 0; t < 8; ++t)
                acc[t] = __builtin_amdgcn_wmma_f32_16x16x32_bf16(
                    false, a, false, b[t], (short)0, acc[t], false, false);
        }
        __syncthreads();                                // all waves done with shA/shW[buf]
        if (c + 1 < NCH && threadIdx.x < 32)
            tdm_load_2d(lds_off(&shA[0]), Ag + (c + 1) * CH, 128);  // safe: single buffer free
    }

    int lane = threadIdx.x & 31, half = lane >> 4, r = lane & 15;
    int rowW0 = row0 + rowLoc;

    int tg[8];
#pragma unroll
    for (int j = 0; j < 8; ++j) {
        int row = rowW0 + half * 8 + j;
        tg[j] = target[(row & (B - 1)) * S + s];
    }
    float sums[8], mxv[8]; int mxi[8];
#pragma unroll
    for (int j = 0; j < 8; ++j) { sums[j] = 0.0f; mxv[j] = -2.0f; mxi[j] = 0x7FFFFFFF; }

#pragma unroll
    for (int t = 0; t < 8; ++t) {
        int col = col0 + t * 16 + r;
        float bl = b_lin[col];
#pragma unroll
        for (int j = 0; j < 8; ++j) {
            float tv = tanhf(acc[t][j] + bl);
            sums[j] += expf(tv);
            if (tv > mxv[j] || (tv == mxv[j] && col < mxi[j])) { mxv[j] = tv; mxi[j] = col; }
            if (col == tg[j]) tgtval[rowW0 + half * 8 + j] = tv;
        }
    }
    // butterfly across 16-lane half-group (xor offsets < 16 keep halves intact)
#pragma unroll
    for (int off = 1; off < 16; off <<= 1) {
#pragma unroll
        for (int j = 0; j < 8; ++j) {
            sums[j] += __shfl_xor(sums[j], off, 32);
            float ov = __shfl_xor(mxv[j], off, 32);
            int   oi = __shfl_xor(mxi[j], off, 32);
            if (ov > mxv[j] || (ov == mxv[j] && oi < mxi[j])) { mxv[j] = ov; mxi[j] = oi; }
        }
    }
#pragma unroll
    for (int j = 0; j < 8; ++j) {
        if (r == j) {
            int row = rowW0 + half * 8 + j;
            atomicAdd(&sumexp[row], sums[j]);
            unsigned int bits = __float_as_uint(mxv[j]);
            unsigned int ok   = (bits & 0x80000000u) ? ~bits : (bits | 0x80000000u);
            unsigned long long key =
                ((unsigned long long)ok << 32) | (unsigned long long)(0xFFFFFFFFu - (unsigned)mxi[j]);
            atomicMax(&argkey[row], key);
        }
    }
}

__global__ __launch_bounds__(256) void k_finalize(const int* __restrict__ target, int s,
                                                  const float* __restrict__ sumexp,
                                                  const unsigned long long* __restrict__ argkey,
                                                  const float* __restrict__ tgtval,
                                                  float* __restrict__ loss,
                                                  int* __restrict__ w_i) {
    int r = blockIdx.x * 256 + threadIdx.x;
    if (r >= R) return;
    int tgt = target[(r & (B - 1)) * S + s];
    float lp = tgtval[r] - logf(sumexp[r]);
    if (tgt != 0) loss[r] += lp;
    w_i[r] = (int)(0xFFFFFFFFu - (unsigned)(argkey[r] & 0xFFFFFFFFull));
}

__global__ __launch_bounds__(256) void k_out(const float* __restrict__ loss,
                                             const float* __restrict__ tdistr,
                                             float* __restrict__ out) {
    int idx = blockIdx.x * 256 + threadIdx.x;
    if (idx >= B * T) return;
    int b = idx / T, t = idx % T;
    out[idx] = loss[t * B + b] * tdistr[idx];   // out[b,t], tdistr is [B,T]
}

// ---------------- host launcher ----------------
extern "C" void kernel_launch(void* const* d_in, const int* in_sizes, int n_in,
                              void* d_out, int out_size, void* d_ws, size_t ws_size,
                              hipStream_t stream) {
    const float* temb   = (const float*)d_in[0];
    const float* wemb   = (const float*)d_in[1];
    const float* tdistr = (const float*)d_in[2];
    const int*   target = (const int*)  d_in[3];
    const float* W_ih   = (const float*)d_in[4];
    const float* W_hh   = (const float*)d_in[5];
    const float* b_ih   = (const float*)d_in[6];
    const float* b_hh   = (const float*)d_in[7];
    const float* W_lin  = (const float*)d_in[8];
    const float* b_lin  = (const float*)d_in[9];
    float* out = (float*)d_out;

    char* ws = (char*)d_ws;
    size_t off = 0;
    auto take = [&](size_t bytes) { void* p = ws + off; off += (bytes + 255) & ~size_t(255); return p; };
    unsigned short* WlinBF = (unsigned short*)take((size_t)V  * H3 * 2); // 98.3 MB (L2-resident)
    unsigned short* WcatBF = (unsigned short*)take((size_t)H4 * H3 * 2); //  6.3 MB
    unsigned short* Abuf   = (unsigned short*)take((size_t)R  * H3 * 2); //  1.5 MB
    float*  gates  = (float*)take((size_t)R * H4 * 4);
    float*  cx     = (float*)take((size_t)R * H  * 4);
    float*  bias   = (float*)take((size_t)H4 * 4);
    float*  loss   = (float*)take((size_t)R * 4);
    float*  sumexp = (float*)take((size_t)R * 4);
    float*  tgtval = (float*)take((size_t)R * 4);
    unsigned long long* argkey = (unsigned long long*)take((size_t)R * 8);
    int*    w_i    = (int*)take((size_t)R * 4);

    {
        long n = (long)V * H3;
        k_convert_wlin<<<dim3((unsigned)((n / 8 + 255) / 256)), 256, 0, stream>>>(W_lin, WlinBF, n);
        k_build_wcat<<<dim3((H4 * H3 + 255) / 256), 256, 0, stream>>>(W_ih, W_hh, b_ih, b_hh, WcatBF, bias);
        k_init<<<dim3((R * H + 255) / 256), 256, 0, stream>>>(temb, Abuf, cx, loss, w_i);
    }

    dim3 blk(256);
    dim3 gridGates(H4 / 128, R / 64);   // 16 x 8
    dim3 gridLogit(V  / 128, R / 128);  // 250 x 4
    dim3 gridRH((R * H + 255) / 256);
    dim3 gridR((R + 255) / 256);

    for (int s = 0; s < S; ++s) {
        k_embed        <<<gridRH,    blk, 0, stream>>>(wemb, w_i, Abuf);
        k_gates_gemm   <<<gridGates, blk, 0, stream>>>(Abuf, WcatBF, gates);
        k_cell         <<<gridRH,    blk, 0, stream>>>(gates, bias, cx, Abuf);
        k_zero_partials<<<gridR,     blk, 0, stream>>>(sumexp, argkey, tgtval);
        k_logits       <<<gridLogit, blk, 0, stream>>>(Abuf, WlinBF, b_lin, target, s,
                                                       sumexp, argkey, tgtval);
        k_finalize     <<<gridR,     blk, 0, stream>>>(target, s, sumexp, argkey, tgtval, loss, w_i);
    }
    k_out<<<gridR, blk, 0, stream>>>(loss, tdistr, out);
}
